// Net_91225105367819
// MI455X (gfx1250) — matrix operand order repaired and checked by reading, not compiled, not run on hardware
//
#include <hip/hip_runtime.h>
#include <hip/hip_bf16.h>
#include <math.h>

typedef __attribute__((ext_vector_type(16))) _Float16 v16h;
typedef __attribute__((ext_vector_type(8)))  _Float16 v8h;
typedef __attribute__((ext_vector_type(8)))  float    v8f;

#define INF_F __builtin_inff()

__device__ __forceinline__ float eluf(float x) { return x > 0.0f ? x : expf(x) - 1.0f; }

// ---------------------------------------------------------------------------
// Encoder: row-per-block 2-layer MLP (in_w -> 128 -> 128, ELU), writes f32,
// f16 copy, and |row|^2 (needed by the fused KNN).
// ---------------------------------------------------------------------------
__global__ void __launch_bounds__(128) encoder_kernel(
    const float* __restrict__ x, int in_w,
    const float* __restrict__ w1, const float* __restrict__ b1,
    const float* __restrict__ w2, const float* __restrict__ b2,
    float* __restrict__ enc, _Float16* __restrict__ ench,
    float* __restrict__ norms)
{
    __shared__ float s_x[8];
    __shared__ float s_h1[128];
    __shared__ float s_red[128];
    const int row = blockIdx.x;
    const int t = threadIdx.x;
    if (t < in_w) s_x[t] = x[(size_t)row * in_w + t];
    __syncthreads();
    float a = b1[t];
    for (int k = 0; k < in_w; ++k) a += s_x[k] * w1[k * 128 + t];
    s_h1[t] = eluf(a);
    __syncthreads();
    float a2 = b2[t];
    #pragma unroll 8
    for (int k = 0; k < 128; ++k) a2 += s_h1[k] * w2[k * 128 + t];
    const float e = eluf(a2);
    enc[(size_t)row * 128 + t] = e;
    ench[(size_t)row * 128 + t] = (_Float16)e;
    s_red[t] = e * e;
    __syncthreads();
    for (int s = 64; s > 0; s >>= 1) {
        if (t < s) s_red[t] += s_red[t + s];
        __syncthreads();
    }
    if (t == 0) norms[row] = s_red[0];
}

// ---------------------------------------------------------------------------
// Row-norm for feats1 (needed by second KNN).
// ---------------------------------------------------------------------------
__global__ void __launch_bounds__(128) rownorm_kernel(
    const float* __restrict__ f, float* __restrict__ norms)
{
    __shared__ float red[128];
    const int row = blockIdx.x, t = threadIdx.x;
    const float v = f[(size_t)row * 128 + t];
    red[t] = v * v;
    __syncthreads();
    for (int s = 64; s > 0; s >>= 1) {
        if (t < s) red[t] += red[t + s];
        __syncthreads();
    }
    if (t == 0) norms[row] = red[0];
}

// ---------------------------------------------------------------------------
// conv_w (256x128 f32, K-major) -> wt (128x256 f16, N-major) so WMMA B tiles
// are 16 contiguous halves along K per lane.
// ---------------------------------------------------------------------------
__global__ void convw_kernel(const float* __restrict__ w, _Float16* __restrict__ wt)
{
    const int i = blockIdx.x * 256 + threadIdx.x;   // 0 .. 256*128-1
    const int k = i >> 7, n = i & 127;
    wt[(size_t)n * 256 + k] = (_Float16)w[i];
}

// ---------------------------------------------------------------------------
// Batch segment boundaries: seg[b] = lower_bound(batch, b), b = 0..16.
// ---------------------------------------------------------------------------
__global__ void seg_kernel(const int* __restrict__ batch, int n, int* __restrict__ seg)
{
    const int b = threadIdx.x;
    if (b > 16) return;
    int lo = 0, hi = n;
    while (lo < hi) {
        const int mid = (lo + hi) >> 1;
        if (batch[mid] < b) lo = mid + 1; else hi = mid;
    }
    seg[b] = lo;
}

// ---------------------------------------------------------------------------
// Fused distance-GEMM + top-8 KNN. One wave per 16 dst rows, 32 src columns
// per iteration (two 16x16 WMMA C tiles). Row norms/batches live in registers
// so masking lowers to v_cmp + v_cndmask (no exec-mask churn). All 32 lanes
// scan: lane owns row lane&15, half-wave picks which 16-column tile; the two
// per-row sorted top-8 lists are merged once at the end through LDS.
// ---------------------------------------------------------------------------
__global__ void __launch_bounds__(32) knn_kernel(
    const _Float16* __restrict__ dsth, const float* __restrict__ dnorm,
    const int* __restrict__ dbatch,
    const _Float16* __restrict__ srch, const float* __restrict__ snorm,
    const int* __restrict__ sbatch,
    const int* __restrict__ seg, int srcN,
    int* __restrict__ outidx)
{
    __shared__ float s_dn[16];
    __shared__ int   s_db[16];
    __shared__ float s_d2[16][33];   // 32 cols + pad (33 coprime to 64 banks)
    __shared__ float s_fd[16][16];   // final merge: per-row 2x8 distances
    __shared__ int   s_fi[16][16];
    const int lane = threadIdx.x;
    const int m0 = blockIdx.x * 16;
    const int half = lane >> 4;
    const int ml = lane & 15;
    if (lane < 16) { s_dn[lane] = dnorm[m0 + lane]; s_db[lane] = dbatch[m0 + lane]; }
    __syncthreads();

    // Per-lane register copies of the 8 C rows this lane covers (m = r + half*8).
    float dn_r[8]; int db_r[8];
    #pragma unroll
    for (int r = 0; r < 8; ++r) { dn_r[r] = s_dn[r + half * 8]; db_r[r] = s_db[r + half * 8]; }

    // A fragments (16 rows x 128 K), contiguous k = 16*half + s per lane (same as B).
    v16h afrag[4];
    const _Float16* abase = dsth + (size_t)(m0 + ml) * 128 + half * 16;
    #pragma unroll
    for (int kc = 0; kc < 4; ++kc) {
        const _Float16* p = abase + kc * 32;
        v16h a;
        *(v8h*)&a       = *(const v8h*)p;         // VGPR0-3: K {0..7} | {16..23}
        *((v8h*)&a + 1) = *(const v8h*)(p + 8);   // VGPR4-7: K {8..15} | {24..31}
        afrag[kc] = a;
    }

    const int b_lo = s_db[0], b_hi = s_db[15];
    const int c0 = seg[b_lo];
    const int c1 = seg[b_hi + 1];

    float d8[8]; int i8[8];
    #pragma unroll
    for (int j = 0; j < 8; ++j) { d8[j] = INF_F; i8[j] = 0; }

    float* d2w = &s_d2[half * 8][ml];            // write base, stride 33

    for (int n0 = (c0 & ~15); n0 < c1; n0 += 32) {
        const int sc0 = n0 + ml;
        const int sc1 = n0 + 16 + ml;
        const int scc0 = sc0 < srcN ? sc0 : srcN - 1;
        const int scc1 = sc1 < srcN ? sc1 : srcN - 1;
        const _Float16* bb0 = srch + (size_t)scc0 * 128 + half * 16;
        const _Float16* bb1 = srch + (size_t)scc1 * 128 + half * 16;
        v8f cA = {}, cB = {};
        #pragma unroll
        for (int kc = 0; kc < 4; ++kc) {
            const v16h b0 = *(const v16h*)(bb0 + kc * 32);
            const v16h b1 = *(const v16h*)(bb1 + kc * 32);
            cA = __builtin_amdgcn_wmma_f32_16x16x32_f16(
                     false, afrag[kc], false, b0, (short)0, cA, false, false);
            cB = __builtin_amdgcn_wmma_f32_16x16x32_f16(
                     false, afrag[kc], false, b1, (short)0, cB, false, false);
        }
        const float sn0 = snorm[scc0], sn1 = snorm[scc1];
        const int   sb0 = sbatch[scc0], sb1 = sbatch[scc1];
        const bool ok0 = (sc0 < srcN), ok1 = (sc1 < srcN);
        #pragma unroll
        for (int r = 0; r < 8; ++r) {
            float dA = fmaf(-2.0f, cA[r], dn_r[r] + sn0);
            float dB = fmaf(-2.0f, cB[r], dn_r[r] + sn1);
            dA = (ok0 && sb0 == db_r[r]) ? dA : INF_F;   // v_cndmask, no branches
            dB = (ok1 && sb1 == db_r[r]) ? dB : INF_F;
            d2w[r * 33]      = dA;                       // col ml
            d2w[r * 33 + 16] = dB;                       // col ml+16
        }
        __syncthreads();
        // Lane owns row ml; its half-wave scans its own 16-column sub-tile.
        const int cb = half * 16;
        #pragma unroll
        for (int c = 0; c < 16; ++c) {
            const float d = s_d2[ml][cb + c];
            if (d < d8[7]) {
                d8[7] = d; i8[7] = n0 + cb + c;
                #pragma unroll
                for (int j = 7; j >= 1; --j) {
                    if (d8[j] < d8[j - 1]) {
                        const float td = d8[j]; d8[j] = d8[j - 1]; d8[j - 1] = td;
                        const int   ti = i8[j]; i8[j] = i8[j - 1]; i8[j - 1] = ti;
                    }
                }
            }
        }
        __syncthreads();
    }

    // Merge the two per-row top-8 lists (one per half-wave) through LDS.
    #pragma unroll
    for (int j = 0; j < 8; ++j) { s_fd[ml][half * 8 + j] = d8[j]; s_fi[ml][half * 8 + j] = i8[j]; }
    __syncthreads();
    if (lane < 16) {
        float f8[8]; int g8[8];
        #pragma unroll
        for (int j = 0; j < 8; ++j) { f8[j] = INF_F; g8[j] = 0; }
        #pragma unroll
        for (int c = 0; c < 16; ++c) {
            const float d = s_fd[lane][c];
            if (d < f8[7]) {
                f8[7] = d; g8[7] = s_fi[lane][c];
                #pragma unroll
                for (int j = 7; j >= 1; --j) {
                    if (f8[j] < f8[j - 1]) {
                        const float td = f8[j]; f8[j] = f8[j - 1]; f8[j - 1] = td;
                        const int   ti = g8[j]; g8[j] = g8[j - 1]; g8[j - 1] = ti;
                    }
                }
            }
        }
        #pragma unroll
        for (int j = 0; j < 8; ++j) outidx[(size_t)(m0 + lane) * 8 + j] = g8[j];
    }
}

// ---------------------------------------------------------------------------
// EdgeConv: 8 dst nodes per block (4 waves). Stage msg = [xi | xj-xi] as
// 64 rows x 256 K f16 in LDS (32 KB). Each wave GEMMs its 16 rows (2 nodes)
// against W (256x128) with WMMA; the C fragment layout puts all 8 neighbor
// rows of one node into one lane's 8 accumulators -> in-register ELU+max.
// ---------------------------------------------------------------------------
__global__ void __launch_bounds__(128) edgeconv_kernel(
    const float* __restrict__ srcf, const float* __restrict__ dstf,
    const int* __restrict__ idx, const _Float16* __restrict__ wt,
    const float* __restrict__ bias,
    float* __restrict__ outf, _Float16* __restrict__ outh)
{
    __shared__ _Float16 s_msg[64 * 256];
    const int tid = threadIdx.x;
    const int g0 = blockIdx.x * 8;            // first dst node of this block
    {
        const int row = tid >> 1;             // 0..63 : node*8 + neighbor
        const int node = row >> 3, jj = row & 7;
        const int gnode = g0 + node;
        const int nbr = idx[(size_t)gnode * 8 + jj];
        const float* xi = dstf + (size_t)gnode * 128;
        const float* xj = srcf + (size_t)nbr * 128;
        const int cbeg = (tid & 1) * 64;
        for (int c = cbeg; c < cbeg + 64; ++c) {
            const float vi = xi[c];
            const float vj = xj[c];
            s_msg[row * 256 + c]       = (_Float16)vi;
            s_msg[row * 256 + 128 + c] = (_Float16)(vj - vi);
        }
    }
    __syncthreads();

    const int wv = tid >> 5, lane = tid & 31;
    const int half = lane >> 4, nl = lane & 15;
    const int lr0 = wv * 16;                  // this wave's first msg row

    v8f acc[8];
    #pragma unroll
    for (int nt = 0; nt < 8; ++nt) acc[nt] = (v8f){};

    const _Float16* abase = s_msg + (lr0 + nl) * 256 + half * 16;
    #pragma unroll
    for (int kc = 0; kc < 8; ++kc) {          // K = 256 in chunks of 32
        const _Float16* pa = abase + kc * 32;
        v16h a;
        *(v8h*)&a       = *(const v8h*)pa;
        *((v8h*)&a + 1) = *(const v8h*)(pa + 8);
        #pragma unroll
        for (int nt = 0; nt < 8; ++nt) {      // N = 128 in tiles of 16
            const _Float16* pb = wt + (size_t)(nt * 16 + nl) * 256 + kc * 32 + half * 16;
            const v16h b = *(const v16h*)pb;
            acc[nt] = __builtin_amdgcn_wmma_f32_16x16x32_f16(
                          false, a, false, b, (short)0, acc[nt], false, false);
        }
    }

    // rows lr0 .. lr0+15 are nodes (2*wv, 2*wv+1); lane half selects the node,
    // accumulator index r is the neighbor index jj.
    const int gnode = g0 + 2 * wv + half;
    #pragma unroll
    for (int nt = 0; nt < 8; ++nt) {
        const int col = nt * 16 + nl;
        const float bb = bias[col];
        float mx = -INF_F;
        #pragma unroll
        for (int r = 0; r < 8; ++r) {
            const float v = eluf(acc[nt][r] + bb);
            mx = v > mx ? v : mx;
        }
        outf[(size_t)gnode * 128 + col] = mx;
        outh[(size_t)gnode * 128 + col] = (_Float16)mx;
    }
}

// ---------------------------------------------------------------------------
// Output head: 128->64->32->4->1 MLP + sigmoid, one block per node.
// Also emits batch_pfc (second tuple element) into d_out's tail.
// ---------------------------------------------------------------------------
__global__ void __launch_bounds__(64) outmlp_kernel(
    const float* __restrict__ f,
    const float* __restrict__ w1, const float* __restrict__ b1,
    const float* __restrict__ w2, const float* __restrict__ b2,
    const float* __restrict__ w3, const float* __restrict__ b3,
    const float* __restrict__ w4, const float* __restrict__ b4,
    const int* __restrict__ batch,
    float* __restrict__ out, int n)
{
    __shared__ float s_in[128];
    __shared__ float s_h1[64];
    __shared__ float s_h2[32];
    __shared__ float s_h3[4];
    const int node = blockIdx.x, t = threadIdx.x;
    s_in[t]      = f[(size_t)node * 128 + t];
    s_in[t + 64] = f[(size_t)node * 128 + 64 + t];
    __syncthreads();
    float a = b1[t];
    #pragma unroll 8
    for (int k = 0; k < 128; ++k) a += s_in[k] * w1[k * 64 + t];
    s_h1[t] = eluf(a);
    __syncthreads();
    if (t < 32) {
        float a2 = b2[t];
        for (int k = 0; k < 64; ++k) a2 += s_h1[k] * w2[k * 32 + t];
        s_h2[t] = eluf(a2);
    }
    __syncthreads();
    if (t < 4) {
        float a3 = b3[t];
        for (int k = 0; k < 32; ++k) a3 += s_h2[k] * w3[k * 4 + t];
        s_h3[t] = eluf(a3);
    }
    __syncthreads();
    if (t == 0) {
        float a4 = b4[0];
        for (int k = 0; k < 4; ++k) a4 += s_h3[k] * w4[k];
        out[node] = 1.0f / (1.0f + expf(-a4));
        ((int*)out)[n + node] = batch[node];   // tuple element 2: batch_pfc
    }
}

// ---------------------------------------------------------------------------
// Host orchestration
// ---------------------------------------------------------------------------
extern "C" void kernel_launch(void* const* d_in, const int* in_sizes, int n_in,
                              void* d_out, int out_size, void* d_ws, size_t ws_size,
                              hipStream_t stream) {
    (void)in_sizes; (void)n_in; (void)out_size; (void)ws_size;
    const int N_PFC = 8192, N_CLUS = 4096;

    const float* x_pfc      = (const float*)d_in[0];
    const float* x_clus     = (const float*)d_in[1];
    const int*   batch_pfc  = (const int*)d_in[3];
    const int*   batch_clus = (const int*)d_in[4];
    const float* pfc_w1  = (const float*)d_in[6];
    const float* pfc_b1  = (const float*)d_in[7];
    const float* pfc_w2  = (const float*)d_in[8];
    const float* pfc_b2  = (const float*)d_in[9];
    const float* clus_w1 = (const float*)d_in[10];
    const float* clus_b1 = (const float*)d_in[11];
    const float* clus_w2 = (const float*)d_in[12];
    const float* clus_b2 = (const float*)d_in[13];
    const float* conv_w  = (const float*)d_in[14];
    const float* conv_b  = (const float*)d_in[15];
    const float* out_w1  = (const float*)d_in[16];
    const float* out_b1  = (const float*)d_in[17];
    const float* out_w2  = (const float*)d_in[18];
    const float* out_b2  = (const float*)d_in[19];
    const float* out_w3  = (const float*)d_in[20];
    const float* out_b3  = (const float*)d_in[21];
    const float* out_w4  = (const float*)d_in[22];
    const float* out_b4  = (const float*)d_in[23];

    char* wsp = (char*)d_ws;
    size_t off = 0;
    auto ALLOC = [&](size_t bytes) -> void* {
        void* p = wsp + off;
        off += (bytes + 255) & ~(size_t)255;
        return p;
    };
    float*     pfc_enc    = (float*)    ALLOC((size_t)N_PFC * 128 * 4);
    _Float16*  pfc_ench   = (_Float16*) ALLOC((size_t)N_PFC * 128 * 2);
    float*     pfc_norm   = (float*)    ALLOC((size_t)N_PFC * 4);
    float*     clus_enc   = (float*)    ALLOC((size_t)N_CLUS * 128 * 4);
    _Float16*  clus_ench  = (_Float16*) ALLOC((size_t)N_CLUS * 128 * 2);
    float*     clus_norm  = (float*)    ALLOC((size_t)N_CLUS * 4);
    float*     feats1     = (float*)    ALLOC((size_t)N_PFC * 128 * 4);
    _Float16*  feats1_h   = (_Float16*) ALLOC((size_t)N_PFC * 128 * 2);
    float*     feats1_n   = (float*)    ALLOC((size_t)N_PFC * 4);
    float*     feats2     = (float*)    ALLOC((size_t)N_PFC * 128 * 4);
    int*       idx1       = (int*)      ALLOC((size_t)N_PFC * 8 * 4);
    int*       idx2       = (int*)      ALLOC((size_t)N_PFC * 8 * 4);
    _Float16*  wt_h       = (_Float16*) ALLOC((size_t)256 * 128 * 2);
    int*       seg_clus   = (int*)      ALLOC(32 * 4);
    int*       seg_pfc    = (int*)      ALLOC(32 * 4);
    _Float16*  feats2_h   = (_Float16*) ALLOC((size_t)N_PFC * 128 * 2); // scratch f16 out

    // Weight transpose/convert + batch segment tables
    convw_kernel<<<128, 256, 0, stream>>>(conv_w, wt_h);
    seg_kernel<<<1, 32, 0, stream>>>(batch_clus, N_CLUS, seg_clus);
    seg_kernel<<<1, 32, 0, stream>>>(batch_pfc,  N_PFC,  seg_pfc);

    // Encoders
    encoder_kernel<<<N_PFC, 128, 0, stream>>>(x_pfc, 8, pfc_w1, pfc_b1, pfc_w2, pfc_b2,
                                              pfc_enc, pfc_ench, pfc_norm);
    encoder_kernel<<<N_CLUS, 128, 0, stream>>>(x_clus, 4, clus_w1, clus_b1, clus_w2, clus_b2,
                                               clus_enc, clus_ench, clus_norm);

    // KNN1: dst = pfc_enc, src = clus_enc
    knn_kernel<<<N_PFC / 16, 32, 0, stream>>>(pfc_ench, pfc_norm, batch_pfc,
                                              clus_ench, clus_norm, batch_clus,
                                              seg_clus, N_CLUS, idx1);
    // EdgeConv1: src = clus_enc, dst = pfc_enc
    edgeconv_kernel<<<N_PFC / 8, 128, 0, stream>>>(clus_enc, pfc_enc, idx1, wt_h, conv_b,
                                                   feats1, feats1_h);
    rownorm_kernel<<<N_PFC, 128, 0, stream>>>(feats1, feats1_n);

    // KNN2: dst = pfc_enc, src = feats1
    knn_kernel<<<N_PFC / 16, 32, 0, stream>>>(pfc_ench, pfc_norm, batch_pfc,
                                              feats1_h, feats1_n, batch_pfc,
                                              seg_pfc, N_PFC, idx2);
    // EdgeConv2: src = feats1, dst = pfc_enc
    edgeconv_kernel<<<N_PFC / 8, 128, 0, stream>>>(feats1, pfc_enc, idx2, wt_h, conv_b,
                                                   feats2, feats2_h);

    // Output head + batch_pfc passthrough
    outmlp_kernel<<<N_PFC, 64, 0, stream>>>(feats2,
                                            out_w1, out_b1, out_w2, out_b2,
                                            out_w3, out_b3, out_w4, out_b4,
                                            batch_pfc, (float*)d_out, N_PFC);
}
